// DTransformer_13297218748633
// MI455X (gfx1250) — matrix operand order repaired and checked by reading, not compiled, not run on hardware
//
#include <hip/hip_runtime.h>
#include <hip/hip_bf16.h>

// ---------------------------------------------------------------------------
// DTransformer forward on MI455X (gfx1250): bf16 WMMA GEMMs + fused
// softmax/distance-decay + LN + broadcast-aware final combine.
// ---------------------------------------------------------------------------

typedef __attribute__((ext_vector_type(16))) __bf16 v16bf;
typedef __attribute__((ext_vector_type(8)))  __bf16 v8bf;
typedef __attribute__((ext_vector_type(8)))  float  v8f;

static constexpr int Bsz = 4;
static constexpr int S   = 1024;
static constexpr int D   = 512;
static constexpr int H   = 8;
static constexpr int DK  = 64;
static constexpr int M_ROWS = Bsz * S;      // 4096

// GEMM tiling
static constexpr int TM = 128;   // block tile M (8 waves x 16 rows)
static constexpr int TN = 64;    // block tile N (4 x 16 subtiles per wave)
static constexpr int TK = 32;    // K step
static constexpr int AP = 48;    // LDS row pitch (bf16 elems) -> 96 B, 32B-aligned
static constexpr int BP = 48;

// ---------------------------------------------------------------------------
// Generic batched WMMA GEMM (templated on B layout):
//   C[z][m][n] = alpha * sum_k A[z][m][k] * B[z][k][n] + bias[n]
// A indexed  A + b*aB + h*aH + m*lda + k              (z = b*heads + h)
// B indexed  (BNMAJOR ? Bsrc[.. + n*ldb + k] : Bsrc[.. + k*ldb + n])
// LDS layouts: As[m][k] (K-contiguous), Bs[n][k] (transposed, K-contiguous)
// so WMMA fragments load as contiguous 16B/32B vectors (ds_load_b128).
// ---------------------------------------------------------------------------
template <int BNMAJOR>
__global__ __launch_bounds__(256)
void wmma_gemm_kernel(const float* __restrict__ A, int lda, long long aB, long long aH,
                      const float* __restrict__ Bsrc, int ldb, long long bB, long long bH,
                      float* __restrict__ C, int ldc, long long cB, long long cH,
                      const float* __restrict__ bias,
                      int K, float alpha, int heads)
{
    __shared__ alignas(32) __bf16 As[TM][AP];   // [m][k]
    __shared__ alignas(32) __bf16 Bs[TN][BP];   // [n][k]  (transposed)

    const int t    = threadIdx.x;
    const int z    = blockIdx.z;
    const int bb   = z / heads;
    const int hh   = z - bb * heads;
    const float* Ap = A    + (long long)bb * aB + (long long)hh * aH;
    const float* Bp = Bsrc + (long long)bb * bB + (long long)hh * bH;
    float*       Cp = C    + (long long)bb * cB + (long long)hh * cH;

    const int gm0 = blockIdx.x * TM;
    const int gn0 = blockIdx.y * TN;

    const int wave = t >> 5;          // 0..7 -> owns rows [wave*16, wave*16+16)
    const int lane = t & 31;
    const int lm   = lane & 15;
    const int half = lane >> 4;

    v8f acc0 = {}, acc1 = {}, acc2 = {}, acc3 = {};

    // staging indices (compile-time shapes)
    const int am  = t >> 1;              // 0..127
    const int akk = (t & 1) * 16;        // 0 / 16
    const int bn  = t >> 2;              // 0..63
    const int bkk = (t & 3) * 8;         // 0,8,16,24
    const int tk  = t >> 3;              // 0..31  (k-major B path)
    const int tn  = (t & 7) * 8;         // 0..56

    for (int k0 = 0; k0 < K; k0 += TK) {
        // ---- stage A tile (128x32): 16 contiguous floats / thread ----
        {
            const float* src = Ap + (long long)(gm0 + am) * lda + (k0 + akk);
            __builtin_prefetch(src + TK, 0, 3);
            const float4* s4 = reinterpret_cast<const float4*>(src);
            float4 f0 = s4[0], f1 = s4[1], f2 = s4[2], f3 = s4[3];
            v16bf tmp;
            tmp[0] = (__bf16)f0.x;  tmp[1] = (__bf16)f0.y;
            tmp[2] = (__bf16)f0.z;  tmp[3] = (__bf16)f0.w;
            tmp[4] = (__bf16)f1.x;  tmp[5] = (__bf16)f1.y;
            tmp[6] = (__bf16)f1.z;  tmp[7] = (__bf16)f1.w;
            tmp[8] = (__bf16)f2.x;  tmp[9] = (__bf16)f2.y;
            tmp[10] = (__bf16)f2.z; tmp[11] = (__bf16)f2.w;
            tmp[12] = (__bf16)f3.x; tmp[13] = (__bf16)f3.y;
            tmp[14] = (__bf16)f3.z; tmp[15] = (__bf16)f3.w;
            *reinterpret_cast<v16bf*>(&As[am][akk]) = tmp;
        }
        // ---- stage B tile (64x32) into transposed Bs[n][k] ----
        if (BNMAJOR) {
            // source W[n][k]: contiguous k read, contiguous k write
            const float* src = Bp + (long long)(gn0 + bn) * ldb + (k0 + bkk);
            __builtin_prefetch(src + TK, 0, 3);
            const float4* s4 = reinterpret_cast<const float4*>(src);
            float4 f0 = s4[0], f1 = s4[1];
            v8bf tmp;
            tmp[0] = (__bf16)f0.x; tmp[1] = (__bf16)f0.y;
            tmp[2] = (__bf16)f0.z; tmp[3] = (__bf16)f0.w;
            tmp[4] = (__bf16)f1.x; tmp[5] = (__bf16)f1.y;
            tmp[6] = (__bf16)f1.z; tmp[7] = (__bf16)f1.w;
            *reinterpret_cast<v8bf*>(&Bs[bn][bkk]) = tmp;
        } else {
            // source V[k][n]: contiguous n read, strided LDS write (transpose)
            const float* src = Bp + (long long)(k0 + tk) * ldb + (gn0 + tn);
            __builtin_prefetch(src + (long long)TK * ldb, 0, 3);
            const float4* s4 = reinterpret_cast<const float4*>(src);
            float4 f0 = s4[0], f1 = s4[1];
            Bs[tn + 0][tk] = (__bf16)f0.x;
            Bs[tn + 1][tk] = (__bf16)f0.y;
            Bs[tn + 2][tk] = (__bf16)f0.z;
            Bs[tn + 3][tk] = (__bf16)f0.w;
            Bs[tn + 4][tk] = (__bf16)f1.x;
            Bs[tn + 5][tk] = (__bf16)f1.y;
            Bs[tn + 6][tk] = (__bf16)f1.z;
            Bs[tn + 7][tk] = (__bf16)f1.w;
        }
        __syncthreads();

        // ---- A fragment: two contiguous 16B runs (ds_load_b128 x2) ----
        const int m = wave * 16 + lm;
        v8bf alo = *reinterpret_cast<const v8bf*>(&As[m][half * 16]);
        v8bf ahi = *reinterpret_cast<const v8bf*>(&As[m][half * 16 + 8]);
        v16bf af = __builtin_shufflevector(alo, ahi,
                      0, 1, 2, 3, 4, 5, 6, 7, 8, 9, 10, 11, 12, 13, 14, 15);

        // ---- 4 N-subtiles, B fragment = contiguous 32B run ----
        v16bf bf0 = *reinterpret_cast<const v16bf*>(&Bs[ 0 + lm][half * 16]);
        v16bf bf1 = *reinterpret_cast<const v16bf*>(&Bs[16 + lm][half * 16]);
        v16bf bf2 = *reinterpret_cast<const v16bf*>(&Bs[32 + lm][half * 16]);
        v16bf bf3 = *reinterpret_cast<const v16bf*>(&Bs[48 + lm][half * 16]);

        acc0 = __builtin_amdgcn_wmma_f32_16x16x32_bf16(false, af, false, bf0, (short)0, acc0, false, false);
        acc1 = __builtin_amdgcn_wmma_f32_16x16x32_bf16(false, af, false, bf1, (short)0, acc1, false, false);
        acc2 = __builtin_amdgcn_wmma_f32_16x16x32_bf16(false, af, false, bf2, (short)0, acc2, false, false);
        acc3 = __builtin_amdgcn_wmma_f32_16x16x32_bf16(false, af, false, bf3, (short)0, acc3, false, false);
        __syncthreads();
    }

    // ---- epilogue: C layout f32 16x16 -> VGPR r holds M=r+8*half, N=lane&15
    const int mBase = gm0 + wave * 16;
#pragma unroll
    for (int s4 = 0; s4 < 4; ++s4) {
        const v8f* accp = (s4 == 0) ? &acc0 : (s4 == 1) ? &acc1 : (s4 == 2) ? &acc2 : &acc3;
        const int nOut = gn0 + s4 * 16 + lm;
        const float bv = bias ? bias[nOut] : 0.0f;
#pragma unroll
        for (int r = 0; r < 8; ++r) {
            const int mOut = mBase + r + 8 * half;
            Cp[(long long)mOut * ldc + nOut] = (*accp)[r] * alpha + bv;
        }
    }
}

// ---------------------------------------------------------------------------
// Fused per-row softmax / distance-decay / second softmax (in-place on SC).
// One 256-thread block per row (row = (b*H+h)*S + i), 4 cols per thread.
// ---------------------------------------------------------------------------
__global__ __launch_bounds__(256)
void softmax_decay_kernel(float* __restrict__ SC, const float* __restrict__ g,
                          int maskK, int zeroPad)
{
    const long long row = blockIdx.x;
    const int i = (int)(row % S);
    const int h = (int)((row / S) % H);
    float* sp = SC + row * (long long)S;

    const int t  = threadIdx.x;
    const int j0 = t * 4;
    __shared__ float red[256];

    float v[4], p[4], c[4];
    {
        const float4 f = *reinterpret_cast<const float4*>(sp + j0);
        v[0] = f.x; v[1] = f.y; v[2] = f.z; v[3] = f.w;
    }

    // ---- masked softmax (mask: j <= i + maskK), fill -1e32 ----
    float mv[4];
    float lmx = -3.0e38f;
#pragma unroll
    for (int q = 0; q < 4; ++q) {
        const bool m = (j0 + q) <= (i + maskK);
        mv[q] = m ? v[q] : -1e32f;
        lmx = fmaxf(lmx, mv[q]);
    }
    red[t] = lmx; __syncthreads();
    for (int off = 128; off > 0; off >>= 1) {
        if (t < off) red[t] = fmaxf(red[t], red[t + off]);
        __syncthreads();
    }
    const float mx1 = red[0]; __syncthreads();

    float lsum = 0.0f;
#pragma unroll
    for (int q = 0; q < 4; ++q) { p[q] = __expf(mv[q] - mx1); lsum += p[q]; }
    red[t] = lsum; __syncthreads();
    for (int off = 128; off > 0; off >>= 1) {
        if (t < off) red[t] += red[t + off];
        __syncthreads();
    }
    const float inv1 = 1.0f / red[0]; __syncthreads();

    // s_ = softmax * mask ; local inclusive cumsum within 4-chunk
    float run = 0.0f;
#pragma unroll
    for (int q = 0; q < 4; ++q) {
        const bool m = (j0 + q) <= (i + maskK);
        p[q] = m ? p[q] * inv1 : 0.0f;
        run += p[q];
        c[q] = run;
    }
    // block inclusive scan of chunk sums
    red[t] = run; __syncthreads();
    for (int off = 1; off < 256; off <<= 1) {
        const float x = (t >= off) ? red[t - off] : 0.0f;
        __syncthreads();
        red[t] += x;
        __syncthreads();
    }
    const float incl  = red[t];
    const float total = red[255];
    __syncthreads();
    const float excl = incl - run;

    // gamma = -softplus(g[h])
    const float gv = g[h];
    const float spl = (gv > 20.0f) ? gv : log1pf(__expf(gv));
    const float gamma = -spl;

    float s2[4];
#pragma unroll
    for (int q = 0; q < 4; ++q) {
        const float distcum = excl + c[q];
        const float pos  = fabsf((float)(i - (j0 + q)));
        const float dist = sqrtf(fmaxf((total - distcum) * pos, 0.0f));
        float eff = __expf(dist * gamma);
        eff = fminf(fmaxf(eff, 1e-5f), 1e5f);
        s2[q] = v[q] * eff;          // NOTE: unmasked raw scores, per reference
    }

    // ---- second softmax (NO mask, faithful to reference no-op masked_fill) ----
    lmx = fmaxf(fmaxf(s2[0], s2[1]), fmaxf(s2[2], s2[3]));
    red[t] = lmx; __syncthreads();
    for (int off = 128; off > 0; off >>= 1) {
        if (t < off) red[t] = fmaxf(red[t], red[t + off]);
        __syncthreads();
    }
    const float mx2 = red[0]; __syncthreads();

    float e2[4]; lsum = 0.0f;
#pragma unroll
    for (int q = 0; q < 4; ++q) { e2[q] = __expf(s2[q] - mx2); lsum += e2[q]; }
    red[t] = lsum; __syncthreads();
    for (int off = 128; off > 0; off >>= 1) {
        if (t < off) red[t] += red[t + off];
        __syncthreads();
    }
    const float inv2 = 1.0f / red[0];

    const bool zero = (zeroPad != 0) && (i == 0);
    float4 o;
    o.x = zero ? 0.0f : e2[0] * inv2;
    o.y = zero ? 0.0f : e2[1] * inv2;
    o.z = zero ? 0.0f : e2[2] * inv2;
    o.w = zero ? 0.0f : e2[3] * inv2;
    *reinterpret_cast<float4*>(sp + j0) = o;
}

// ---------------------------------------------------------------------------
// y = LayerNorm(res + x) * scale + bias  over last dim D=512; one block/row.
// ---------------------------------------------------------------------------
__global__ __launch_bounds__(256)
void add_ln_kernel(const float* __restrict__ res, const float* __restrict__ x,
                   const float* __restrict__ scale, const float* __restrict__ bias,
                   float* __restrict__ out)
{
    const int r = blockIdx.x;
    const int t = threadIdx.x;
    __shared__ float red[256];

    const float* xr = x   + (long long)r * D;
    const float* rr = res + (long long)r * D;
    const float a0 = rr[t] + xr[t];
    const float a1 = rr[t + 256] + xr[t + 256];

    red[t] = a0 + a1; __syncthreads();
    for (int off = 128; off > 0; off >>= 1) {
        if (t < off) red[t] += red[t + off];
        __syncthreads();
    }
    const float mean = red[0] * (1.0f / D); __syncthreads();

    const float d0 = a0 - mean, d1 = a1 - mean;
    red[t] = d0 * d0 + d1 * d1; __syncthreads();
    for (int off = 128; off > 0; off >>= 1) {
        if (t < off) red[t] += red[t + off];
        __syncthreads();
    }
    const float var = red[0] * (1.0f / D);
    const float inv = rsqrtf(var + 1e-5f);

    float* orow = out + (long long)r * D;
    orow[t]       = d0 * inv * scale[t]       + bias[t];
    orow[t + 256] = d1 * inv * scale[t + 256] + bias[t + 256];
}

// ---------------------------------------------------------------------------
// Broadcast know[D] to QUERY[M_ROWS, D]
// ---------------------------------------------------------------------------
__global__ __launch_bounds__(256)
void broadcast_kernel(const float* __restrict__ know, float* __restrict__ out, int n)
{
    const int idx = blockIdx.x * 256 + threadIdx.x;
    if (idx < n) out[idx] = know[idx & (D - 1)];
}

// ---------------------------------------------------------------------------
// kt[h][d] = sigmoid( know[h*64 .. ] . Wlk[d, :] + b_lk[d] )   (8 x 512)
// (query is a broadcast of `know`, so key_t is (b,s)-invariant)
// ---------------------------------------------------------------------------
__global__ __launch_bounds__(256)
void kt_kernel(const float* __restrict__ know, const float* __restrict__ Wlk,
               const float* __restrict__ blk, float* __restrict__ KT)
{
    const int idx = blockIdx.x * 256 + threadIdx.x;
    if (idx >= H * D) return;
    const int h = idx >> 9;
    const int d = idx & (D - 1);
    float s = blk[d];
#pragma unroll 8
    for (int k = 0; k < DK; ++k) s += know[h * DK + k] * Wlk[d * DK + k];
    KT[idx] = 1.0f / (1.0f + __expf(-s));
}

// ---------------------------------------------------------------------------
// Final combine per (b,s) row:
//   beta_h = kt[h,:] . q_emb_row ; alpha = softmax_h(beta)
//   out[d] = sum_h alpha_h * sigmoid( h_row[h*64:] . Wlv[d,:] + b_lv[d] )
// ---------------------------------------------------------------------------
__global__ __launch_bounds__(256)
void combine_kernel(const float* __restrict__ qe, const float* __restrict__ hh,
                    const float* __restrict__ KT, const float* __restrict__ Wlv,
                    const float* __restrict__ blv, float* __restrict__ out)
{
    const int r = blockIdx.x;
    const int t = threadIdx.x;
    __shared__ float qrow[D], hrow[D], red[256];

    qrow[t] = qe[(long long)r * D + t];
    qrow[t + 256] = qe[(long long)r * D + t + 256];
    hrow[t] = hh[(long long)r * D + t];
    hrow[t + 256] = hh[(long long)r * D + t + 256];
    __syncthreads();

    float beta[H];
#pragma unroll
    for (int h8 = 0; h8 < H; ++h8) {
        const float s = KT[h8 * D + t] * qrow[t] + KT[h8 * D + t + 256] * qrow[t + 256];
        red[t] = s; __syncthreads();
        for (int off = 128; off > 0; off >>= 1) {
            if (t < off) red[t] += red[t + off];
            __syncthreads();
        }
        beta[h8] = red[0];
        __syncthreads();
    }
    // softmax over 8 heads (tiny; every thread replicates)
    float bm = beta[0];
#pragma unroll
    for (int h8 = 1; h8 < H; ++h8) bm = fmaxf(bm, beta[h8]);
    float es = 0.0f, a[H];
#pragma unroll
    for (int h8 = 0; h8 < H; ++h8) { a[h8] = __expf(beta[h8] - bm); es += a[h8]; }
    const float invs = 1.0f / es;
#pragma unroll
    for (int h8 = 0; h8 < H; ++h8) a[h8] *= invs;

    for (int dd = t; dd < D; dd += 256) {
        float acc = 0.0f;
        const float* wp = Wlv + (long long)dd * DK;
        const float  bb = blv[dd];
#pragma unroll
        for (int h8 = 0; h8 < H; ++h8) {
            float s = bb;
            const float* hp = hrow + h8 * DK;
#pragma unroll 8
            for (int k = 0; k < DK; ++k) s += hp[k] * wp[k];
            acc += a[h8] * (1.0f / (1.0f + __expf(-s)));
        }
        out[(long long)r * D + dd] = acc;
    }
}

// ---------------------------------------------------------------------------
// Host-side orchestration
// ---------------------------------------------------------------------------
static void launch_proj(const float* X, const float* W, const float* b, float* Y,
                        hipStream_t s)
{
    dim3 g(M_ROWS / TM, D / TN, 1);
    wmma_gemm_kernel<1><<<g, 256, 0, s>>>(X, D, 0, 0,
                                          W, D, 0, 0,
                                          Y, D, 0, 0,
                                          b, /*K=*/D, 1.0f, 1);
}

static void launch_scores(const float* Q, const float* Kp, float* SC, hipStream_t s)
{
    dim3 g(S / TM, S / TN, Bsz * H);
    wmma_gemm_kernel<1><<<g, 256, 0, s>>>(Q, D, (long long)S * D, DK,
                                          Kp, D, (long long)S * D, DK,
                                          SC, S, (long long)H * S * S, (long long)S * S,
                                          nullptr, /*K=*/DK, 0.125f, H);
}

static void launch_attv(const float* SC, const float* V, float* CTX, hipStream_t s)
{
    dim3 g(S / TM, 1, Bsz * H);   // N=64 -> gy=1
    wmma_gemm_kernel<0><<<g, 256, 0, s>>>(SC, S, (long long)H * S * S, (long long)S * S,
                                          V, D, (long long)S * D, DK,
                                          CTX, D, (long long)S * D, DK,
                                          nullptr, /*K=*/S, 1.0f, H);
}

extern "C" void kernel_launch(void* const* d_in, const int* in_sizes, int n_in,
                              void* d_out, int out_size, void* d_ws, size_t ws_size,
                              hipStream_t stream)
{
    (void)in_sizes; (void)n_in; (void)out_size; (void)ws_size;

    const float* q_emb  = (const float*)d_in[0];
    const float* qa_emb = (const float*)d_in[1];
    const float* Wq1 = (const float*)d_in[2];  const float* bq1 = (const float*)d_in[3];
    const float* Wv1 = (const float*)d_in[4];  const float* bv1 = (const float*)d_in[5];
    const float* Wo1 = (const float*)d_in[6];  const float* bo1 = (const float*)d_in[7];
    const float* g1  = (const float*)d_in[8];
    const float* ls1 = (const float*)d_in[9];  const float* lb1 = (const float*)d_in[10];
    const float* Wq2 = (const float*)d_in[11]; const float* bq2 = (const float*)d_in[12];
    const float* Wv2 = (const float*)d_in[13]; const float* bv2 = (const float*)d_in[14];
    const float* Wo2 = (const float*)d_in[15]; const float* bo2 = (const float*)d_in[16];
    const float* g2  = (const float*)d_in[17];
    const float* ls2 = (const float*)d_in[18]; const float* lb2 = (const float*)d_in[19];
    const float* Wq3 = (const float*)d_in[20]; const float* bq3 = (const float*)d_in[21];
    const float* Wk3 = (const float*)d_in[22]; const float* bk3 = (const float*)d_in[23];
    const float* Wv3 = (const float*)d_in[24]; const float* bv3 = (const float*)d_in[25];
    const float* Wo3 = (const float*)d_in[26]; const float* bo3 = (const float*)d_in[27];
    const float* g3  = (const float*)d_in[28];
    const float* ls3 = (const float*)d_in[29]; const float* lb3 = (const float*)d_in[30];
    const float* know = (const float*)d_in[31];
    const float* Wlk  = (const float*)d_in[32]; const float* b_lk = (const float*)d_in[33];
    const float* Wlv  = (const float*)d_in[34]; const float* b_lv = (const float*)d_in[35];

    // workspace layout (floats)
    const long long ND = (long long)M_ROWS * D;       // 2M floats
    float* W0 = (float*)d_ws;
    float* Qb    = W0 + 0 * ND;
    float* Vb    = W0 + 1 * ND;
    float* CTX   = W0 + 2 * ND;
    float* MO    = W0 + 3 * ND;
    float* HQ    = W0 + 4 * ND;
    float* HA    = W0 + 5 * ND;
    float* HH    = W0 + 6 * ND;
    float* QUERY = W0 + 7 * ND;
    float* KP    = W0 + 8 * ND;
    float* SC    = W0 + 9 * ND;                       // B*H*S*S = 32M floats
    float* KT    = SC + (long long)Bsz * H * S * S;   // 8*512

    const int rows_sm = Bsz * H * S;                  // softmax rows

    // ---------------- block 1 (self-attn on q_emb, kq_same) ----------------
    launch_proj(q_emb, Wq1, bq1, Qb, stream);
    launch_proj(q_emb, Wv1, bv1, Vb, stream);
    launch_scores(Qb, Qb, SC, stream);
    softmax_decay_kernel<<<rows_sm, 256, 0, stream>>>(SC, g1, /*maskK=*/0, /*zeroPad=*/0);
    launch_attv(SC, Vb, CTX, stream);
    launch_proj(CTX, Wo1, bo1, MO, stream);
    add_ln_kernel<<<M_ROWS, 256, 0, stream>>>(q_emb, MO, ls1, lb1, HQ);

    // ---------------- block 2 (self-attn on qa_emb, kq_same) ---------------
    launch_proj(qa_emb, Wq2, bq2, Qb, stream);
    launch_proj(qa_emb, Wv2, bv2, Vb, stream);
    launch_scores(Qb, Qb, SC, stream);
    softmax_decay_kernel<<<rows_sm, 256, 0, stream>>>(SC, g2, 0, 0);
    launch_attv(SC, Vb, CTX, stream);
    launch_proj(CTX, Wo2, bo2, MO, stream);
    add_ln_kernel<<<M_ROWS, 256, 0, stream>>>(qa_emb, MO, ls2, lb2, HA);

    // ---------------- block 3 (cross: query=know, k=hq, v=ha) --------------
    broadcast_kernel<<<(int)((ND + 255) / 256), 256, 0, stream>>>(know, QUERY, (int)ND);
    launch_proj(QUERY, Wq3, bq3, Qb, stream);
    launch_proj(HQ, Wk3, bk3, KP, stream);
    launch_proj(HA, Wv3, bv3, Vb, stream);
    launch_scores(Qb, KP, SC, stream);
    softmax_decay_kernel<<<rows_sm, 256, 0, stream>>>(SC, g3, /*maskK=*/-1, /*zeroPad=*/1);
    launch_attv(SC, Vb, CTX, stream);
    launch_proj(CTX, Wo3, bo3, MO, stream);
    add_ln_kernel<<<M_ROWS, 256, 0, stream>>>(QUERY, MO, ls3, lb3, HH);

    // ---------------- final combine ----------------------------------------
    kt_kernel<<<(H * D + 255) / 256, 256, 0, stream>>>(know, Wlk, b_lk, KT);
    combine_kernel<<<M_ROWS, 256, 0, stream>>>(q_emb, HH, KT, Wlv, b_lv, (float*)d_out);
}
